// GaugeNet_52020643889789
// MI455X (gfx1250) — matrix operand order repaired and checked by reading, not compiled, not run on hardware
//
#include <hip/hip_runtime.h>
#include <hip/hip_bf16.h>

typedef __attribute__((ext_vector_type(16))) _Float16 v16h;
typedef __attribute__((ext_vector_type(8)))  float    v8f;
typedef __attribute__((ext_vector_type(2)))  float    v2f;

namespace {
constexpr int kGrid   = 128;
constexpr int kSites  = kGrid * kGrid;              // 16384
constexpr int kHid    = 64;
constexpr int kBatch  = 64;
constexpr int kSplits = 16;                         // blocks per batch
constexpr int kThreads = 256;                       // 8 wave32 waves
constexpr int kWaves  = kThreads / 32;
constexpr int kSitesPerBlock = kSites / kSplits;    // 1024
constexpr int kTilesPerBlock = kSitesPerBlock / 16; // 64
constexpr int kTilesPerWave  = kTilesPerBlock / kWaves; // 8
}

// Single-instruction relu: v_med3_f32(x, 0, +inf)
__device__ __forceinline__ float relu1(float v) {
    return __builtin_amdgcn_fmed3f(v, 0.0f, __builtin_inff());
}

__global__ void gauge_init(float* out, const float* __restrict__ b_post) {
    int b = blockIdx.x * blockDim.x + threadIdx.x;
    if (b < kBatch) out[b] = (float)kSites * b_post[0];
}

__global__ __launch_bounds__(kThreads)
void gauge_net(const float* __restrict__ x,
               const float* __restrict__ H,
               const float* __restrict__ W_emb,
               const float* __restrict__ b_emb,
               const float* __restrict__ W_hid,
               const float* __restrict__ b_hid,
               const float* __restrict__ W_post,
               float* __restrict__ out) {
    const int tid  = threadIdx.x;
    const int lane = tid & 31;
    const int wave = tid >> 5;
    const int half = lane >> 4;
    const int l15  = lane & 15;

    const int b     = blockIdx.x / kSplits;
    const int split = blockIdx.x % kSplits;
    const float* xb = x + (size_t)b * kSites * 2;

    // ---- loop-invariant operand preload (all kept in VGPRs) ----
    v2f  aW[4];      // W_emb^T tiles in f32 16x4 A layout
    v8f  cA[4];      // b_emb in C layout of the transposed step-A output
    v16h bh[2][4];   // W_hid in f16 32x16 B layout: [K chunk][col block]
    float cHv[4];    // b_hid per col block (broadcast along rows)
    float wp[4];     // W_post per col block
#pragma unroll
    for (int c = 0; c < 4; ++c) {
        aW[c].x = W_emb[(2 * half + 0) * kHid + 16 * c + l15];
        aW[c].y = W_emb[(2 * half + 1) * kHid + 16 * c + l15];
#pragma unroll
        for (int j = 0; j < 8; ++j)
            cA[c][j] = b_emb[16 * c + 8 * half + j];
#pragma unroll
        for (int kc = 0; kc < 2; ++kc) {
#pragma unroll
            for (int e = 0; e < 16; ++e) {
                int kg = 32 * kc + 8 * half + e + (e >= 8 ? 8 : 0);     // B: half0 K=0..7,16..23, half1 K=8..15,24..31
                bh[kc][c][e] = (_Float16)W_hid[kg * kHid + 16 * c + l15];
            }
        }
        cHv[c] = b_hid[16 * c + l15];
        wp[c]  = W_post[16 * c + l15];
    }
    const float H00 = H[0], H01 = H[1], H10 = H[2], H11 = H[3];

    float p = 0.0f;

    for (int t = 0; t < kTilesPerWave; ++t) {
        const int site_base = split * kSitesPerBlock + (wave * kTilesPerWave + t) * 16;
        const int site = site_base + l15;
        const int xc = site & (kGrid - 1);
        const int yc = site >> 7;
        const int up = xc + (((yc + 1) & (kGrid - 1)) << 7);
        const int dn = xc + (((yc + kGrid - 1) & (kGrid - 1)) << 7);
        const int lf = ((xc + kGrid - 1) & (kGrid - 1)) + (yc << 7);
        const int rt = ((xc + 1) & (kGrid - 1)) + (yc << 7);
        const int nA = half ? lf : up;   // B' row pair: half0 -> k=0,1 ; half1 -> k=2,3
        const int nB = half ? rt : dn;

        const float x0 = xb[2 * site + 0];
        const float x1 = xb[2 * site + 1];
        const float y0 = x0 * H00 + x1 * H10;   // y_n = sum_m x_m H[m,n]
        const float y1 = x0 * H01 + x1 * H11;

        v2f sv;                                  // s^T in f32 4x16 B layout
        sv.x = y0 * xb[2 * nA + 0] + y1 * xb[2 * nA + 1];
        sv.y = y0 * xb[2 * nB + 0] + y1 * xb[2 * nB + 1];

        // Step A: h1^T blocks = W_emb^T(16x4) x s^T(4x16) + b_emb, then relu.
        v8f d1[4];
#pragma unroll
        for (int c = 0; c < 4; ++c) {
            d1[c] = __builtin_amdgcn_wmma_f32_16x16x4_f32(
                false, aW[c], false, sv, (short)0, cA[c], false, false);
#pragma unroll
            for (int j = 0; j < 8; ++j)
                d1[c][j] = relu1(d1[c][j]);
        }

        // Transposed D' blocks are exactly the f16 16x32 A-operand layout:
        // pack to two K-chunks, pure register work (no LDS transpose).
        v16h a0, a1;
#pragma unroll
        for (int j = 0; j < 8; ++j) {
            a0[j]     = (_Float16)d1[0][j];
            a0[j + 8] = (_Float16)d1[1][j];
            a1[j]     = (_Float16)d1[2][j];
            a1[j + 8] = (_Float16)d1[3][j];
        }

        // Step B: h2 = relu(h1 @ W_hid + b_hid); epilogue dot with W_post.
#pragma unroll
        for (int c = 0; c < 4; ++c) {
            v8f acc;
#pragma unroll
            for (int j = 0; j < 8; ++j) acc[j] = cHv[c];
            acc = __builtin_amdgcn_wmma_f32_16x16x32_f16(
                false, a0, false, bh[0][c], (short)0, acc, false, false);
            acc = __builtin_amdgcn_wmma_f32_16x16x32_f16(
                false, a1, false, bh[1][c], (short)0, acc, false, false);
#pragma unroll
            for (int j = 0; j < 8; ++j)
                p += relu1(acc[j]) * wp[c];
        }
    }

    __shared__ float red[kThreads];
    red[tid] = p;
    __syncthreads();
#pragma unroll
    for (int s = kThreads / 2; s > 0; s >>= 1) {
        if (tid < s) red[tid] += red[tid + s];
        __syncthreads();
    }
    if (tid == 0) atomicAdd(&out[b], red[0]);
}

extern "C" void kernel_launch(void* const* d_in, const int* in_sizes, int n_in,
                              void* d_out, int out_size, void* d_ws, size_t ws_size,
                              hipStream_t stream) {
    const float* x      = (const float*)d_in[0];
    const float* H      = (const float*)d_in[1];
    const float* W_emb  = (const float*)d_in[2];
    const float* b_emb  = (const float*)d_in[3];
    const float* W_hid  = (const float*)d_in[4];
    const float* b_hid  = (const float*)d_in[5];
    const float* W_post = (const float*)d_in[6];
    const float* b_post = (const float*)d_in[7];
    float* out = (float*)d_out;

    gauge_init<<<1, 64, 0, stream>>>(out, b_post);
    gauge_net<<<dim3(kBatch * kSplits), dim3(kThreads), 0, stream>>>(
        x, H, W_emb, b_emb, W_hid, b_hid, W_post, out);
}
